// SchNet_5806795784377
// MI455X (gfx1250) — matrix operand order repaired and checked by reading, not compiled, not run on hardware
//
#include <hip/hip_runtime.h>

typedef __attribute__((ext_vector_type(16))) _Float16 v16h;
typedef __attribute__((ext_vector_type(8)))  _Float16 v8h;
typedef __attribute__((ext_vector_type(8)))  float    v8f;

static constexpr int Bb = 64, NN = 16, NE = 64, K = 64, D = 128, F = 32;
static constexpr int NNODES  = Bb * (NN + NE);   // 5120
static constexpr int NELEC   = Bb * NE;          // 4096
static constexpr int ZSTRIDE = NNODES * K;       // one z table (floats)

__device__ __forceinline__ float ssp(float x) {
  // softplus(x) - ln2, stable form
  return fmaxf(x, 0.0f) + __logf(1.0f + __expf(-fabsf(x))) - 0.69314718055994530942f;
}

// wave-local LDS producer->consumer sync (per-wave scratch only):
// all lanes' ds_stores retired before any subsequent ds_load issues.
__device__ __forceinline__ void wave_lds_sync() {
  __builtin_amdgcn_wave_barrier();
  asm volatile("s_wait_dscnt 0x0" ::: "memory");
  __builtin_amdgcn_wave_barrier();
}

// ---- WMMA fragment layouts (CDNA5 ISA 7.12.2, wave32) ----
// 16-bit A 16x32: lane halves 0..7 -> k = 8*hi .. 8*hi+7,
//                 halves 8..15    -> k = 16+8*hi .. 16+8*hi+7   (hi = lane>>4)
// 16-bit B 32x16: same k order as A, n = lane&15 (2/VGPR)

__device__ __forceinline__ v16h pack_a(float4 x0, float4 x1, float4 x2,
                                       float4 x3) {
  v16h a;
  a[0]  = (_Float16)x0.x; a[1]  = (_Float16)x0.y;
  a[2]  = (_Float16)x0.z; a[3]  = (_Float16)x0.w;
  a[4]  = (_Float16)x1.x; a[5]  = (_Float16)x1.y;
  a[6]  = (_Float16)x1.z; a[7]  = (_Float16)x1.w;
  a[8]  = (_Float16)x2.x; a[9]  = (_Float16)x2.y;
  a[10] = (_Float16)x2.z; a[11] = (_Float16)x2.w;
  a[12] = (_Float16)x3.x; a[13] = (_Float16)x3.y;
  a[14] = (_Float16)x3.z; a[15] = (_Float16)x3.w;
  return a;
}

// A fragment from row-major f32: two contiguous 32B runs per lane -> 4x b128
__device__ __forceinline__ v16h load_a_vec(const float* __restrict__ src, int ld) {
  const int lane = threadIdx.x & 31;
  const int m = lane & 15, hi = lane >> 4;
  const float* r = src + (size_t)m * ld + 8 * hi;
  return pack_a(*(const float4*)(r), *(const float4*)(r + 4),
                *(const float4*)(r + 16), *(const float4*)(r + 20));
}

// Cooperative stage of a (KT*32 x NT*16) f32 weight into fragment-ordered f16
// LDS: lane L's 16 halves of tile (kt*NT+nt) are contiguous (32B).
__device__ __forceinline__ void stage_bfrags(const float* __restrict__ W, int ldn,
                                             int KT, int NT,
                                             _Float16* __restrict__ dst) {
  const int total = KT * NT * 512;
  for (int i = threadIdx.x; i < total; i += blockDim.x) {
    const int half = i & 15;
    const int lane = (i >> 4) & 31;
    const int tile = i >> 9;                 // kt*NT + nt
    const int nt = tile % NT, kt = tile / NT;
    const int hi = lane >> 4;
    const int k = kt * 32 + 8 * hi + (half & 7) + 16 * (half >> 3);
    const int n = nt * 16 + (lane & 15);
    dst[i] = (_Float16)W[(size_t)k * ldn + n];
  }
}

// single 32B LDS read per B fragment
__device__ __forceinline__ v16h fetch_bfrag(const _Float16* __restrict__ frags,
                                            int tile) {
  const int lane = threadIdx.x & 31;
  return *(const v16h*)(frags + ((size_t)tile * 32 + lane) * 16);
}

__device__ __forceinline__ v8f wmma16(v16h a, v16h b, v8f c) {
  return __builtin_amdgcn_wmma_f32_16x16x32_f16(false, a, false, b, (short)0, c,
                                                false, false);
}

// -------------------- kernels --------------------

__global__ void zero_kernel(float* __restrict__ p, int n) {
  int i = blockIdx.x * blockDim.x + threadIdx.x;
  const int stride = gridDim.x * blockDim.x;
  for (; i < n; i += stride) p[i] = 0.0f;
}

// bucket edge ids by type (0: et==3 same, 1: et==4 anti, 2: et==1 nuc)
// warp-aggregated atomics: one atomicAdd per wave per type.
__global__ void compact_kernel(const int* __restrict__ e_typ, int E,
                               int* __restrict__ cnt, int* __restrict__ lists) {
  const int lane = threadIdx.x & 31;
  int i = blockIdx.x * blockDim.x + threadIdx.x;
  const int stride = gridDim.x * blockDim.x;
  for (; i < E; i += stride) {   // E % 32 == 0 -> wave-uniform trip count
    const int et = e_typ[i];
    const int t = (et == 3) ? 0 : (et == 4) ? 1 : (et == 1) ? 2 : -1;
#pragma unroll
    for (int tt = 0; tt < 3; ++tt) {
      const unsigned long long m = __ballot(t == tt);
      if (m != 0ull) {
        const int leader = __ffsll((long long)m) - 1;
        int pbase = 0;
        if (lane == leader) pbase = atomicAdd(&cnt[tt], __popcll(m));
        pbase = __shfl(pbase, leader);
        if (t == tt) {
          const int off = __popcll(m & ((1ull << lane) - 1ull));
          lists[(size_t)tt * E + pbase + off] = i;
        }
      }
    }
  }
}

// hx[node][64]: nuc rows copied, elec rows = elec @ h_W + h_b (WMMA)
__global__ void __launch_bounds__(128) hx_kernel(
    const float* __restrict__ nuc, const float* __restrict__ elec,
    const float* __restrict__ hW, const float* __restrict__ hb,
    float* __restrict__ hx) {
  const int GEMM_BLOCKS = NELEC / 16 / 4;  // 64 blocks * 4 waves = 256 tiles
  __shared__ alignas(32) _Float16 sHW[4 * 4 * 512];  // 4 K-steps x 4 N-tiles
  const int lane = threadIdx.x & 31;
  const int wid  = threadIdx.x >> 5;
  if (blockIdx.x < GEMM_BLOCKS) {
    stage_bfrags(hW, K, 4, 4, sHW);
    __syncthreads();
    const int wt = blockIdx.x * 4 + wid;   // 16 elec rows per wave
    const int r0 = wt * 16;
    v8f zero = {};
    v8f acc[4] = {zero, zero, zero, zero};
#pragma unroll
    for (int kt = 0; kt < 4; ++kt) {       // D=128 -> 4 K-steps of 32
      v16h a = load_a_vec(elec + (size_t)r0 * D + kt * 32, D);
#pragma unroll
      for (int nt = 0; nt < 4; ++nt)       // K=64 -> 4 N-tiles
        acc[nt] = wmma16(a, fetch_bfrag(sHW, kt * 4 + nt), acc[nt]);
    }
    const int n0 = lane & 15, hi = lane >> 4;
#pragma unroll
    for (int nt = 0; nt < 4; ++nt) {
      const int n = nt * 16 + n0;
#pragma unroll
      for (int v = 0; v < 8; ++v) {
        const int r = r0 + v + 8 * hi;
        const int node = (r >> 6) * (NN + NE) + NN + (r & 63);
        hx[(size_t)node * K + n] = acc[nt][v] + hb[n];
      }
    }
  } else {  // copy nuc rows into hx (vectorized b128)
    const float4* src = (const float4*)nuc;
    int i = (blockIdx.x - GEMM_BLOCKS) * blockDim.x + threadIdx.x;
    const int total  = Bb * NN * K / 4;
    const int stride = (gridDim.x - GEMM_BLOCKS) * blockDim.x;
    const int rowq = K / 4;  // 16 float4 per row
    for (; i < total; i += stride) {
      const int b = i / (NN * rowq);
      const int rem = i - b * NN * rowq;
      const int row = rem / rowq, kq = rem % rowq;
      *(float4*)(hx + (((size_t)b * (NN + NE) + row) * K) + kq * 4) = src[i];
    }
  }
}

// grid-stride over type-homogeneous 16-edge tiles from the compacted lists:
// one MLP per tile, f32-atomic scatter into z[type][receiver][64]
__global__ void __launch_bounds__(128) edge_kernel(
    const float* __restrict__ dist,
    const float* __restrict__ W1_same, const float* __restrict__ b1_same,
    const float* __restrict__ W2_same, const float* __restrict__ b2_same,
    const float* __restrict__ W1_anti, const float* __restrict__ b1_anti,
    const float* __restrict__ W2_anti, const float* __restrict__ b2_anti,
    const float* __restrict__ W1_n,    const float* __restrict__ b1_n,
    const float* __restrict__ W2_n,    const float* __restrict__ b2_n,
    const int* __restrict__ senders, const int* __restrict__ receivers,
    const float* __restrict__ hx, float* __restrict__ z,
    const int* __restrict__ cnt, const int* __restrict__ lists, int E) {
  __shared__ alignas(32) _Float16 sW1f[3][2 * 512];   // 2 N-tiles per type
  __shared__ alignas(32) _Float16 sW2f[3][4 * 512];   // 4 N-tiles per type
  __shared__ float    sB1[3][F];
  __shared__ float    sB2[3][K];
  __shared__ alignas(32) _Float16 sH[4][16 * F];      // per-wave H (16x32 f16)

  const float* W1g[3] = {W1_same, W1_anti, W1_n};
  const float* W2g[3] = {W2_same, W2_anti, W2_n};
  const float* b1g[3] = {b1_same, b1_anti, b1_n};
  const float* b2g[3] = {b2_same, b2_anti, b2_n};

#pragma unroll
  for (int t = 0; t < 3; ++t) {
    stage_bfrags(W1g[t], F, 1, 2, sW1f[t]);
    stage_bfrags(W2g[t], K, 1, 4, sW2f[t]);
  }
  for (int i = threadIdx.x; i < 3 * F; i += blockDim.x)
    sB1[i / F][i % F] = b1g[i / F][i % F];
  for (int i = threadIdx.x; i < 3 * K; i += blockDim.x)
    sB2[i / K][i % K] = b2g[i / K][i % K];
  __syncthreads();

  const int lane = threadIdx.x & 31;
  const int wid  = threadIdx.x >> 5;
  const int n0 = lane & 15, hi = lane >> 4;
  const int nwaves = gridDim.x * 4;

  const int c0 = cnt[0], c1 = cnt[1], c2 = cnt[2];
  const int t0 = (c0 + 15) >> 4, t1 = (c1 + 15) >> 4, t2 = (c2 + 15) >> 4;
  const int total_tiles = t0 + t1 + t2;
  _Float16* Hw = sH[wid];

  for (int gt = blockIdx.x * 4 + wid; gt < total_tiles; gt += nwaves) {
    int t, tt;
    if (gt < t0)            { t = 0; tt = gt; }
    else if (gt < t0 + t1)  { t = 1; tt = gt - t0; }
    else                    { t = 2; tt = gt - t0 - t1; }
    const int ct = (t == 0) ? c0 : (t == 1) ? c1 : c2;
    const int base = tt * 16;
    const int* lst = lists + (size_t)t * E;

    // A fragment: 16 same-type edge rows; per-lane row base from list
    const int myidx = base + n0;
    const int erow = lst[(myidx < ct) ? myidx : (ct - 1)];
    const float* r = dist + (size_t)erow * F + 8 * hi;
    v16h a = pack_a(*(const float4*)(r), *(const float4*)(r + 4),
                    *(const float4*)(r + 16), *(const float4*)(r + 20));

    // H = ssp(dist @ W1 + b1)  (16x32), f16 to wave-private LDS
    v8f c0v = {}, c1v = {};
    c0v = wmma16(a, fetch_bfrag(sW1f[t], 0), c0v);
    c1v = wmma16(a, fetch_bfrag(sW1f[t], 1), c1v);
#pragma unroll
    for (int v = 0; v < 8; ++v) {
      const int row = v + 8 * hi;
      Hw[row * F + n0]      = (_Float16)ssp(c0v[v] + sB1[t][n0]);
      Hw[row * F + 16 + n0] = (_Float16)ssp(c1v[v] + sB1[t][16 + n0]);
    }
    wave_lds_sync();  // C-layout -> A-layout reshuffle (wave-private)

    const _Float16* Hr = Hw + n0 * F;
    v8h l0 = *(const v8h*)(Hr + 8 * hi);
    v8h l1 = *(const v8h*)(Hr + 16 + 8 * hi);
    v16h ah = __builtin_shufflevector(l0, l1, 0, 1, 2, 3, 4, 5, 6, 7, 8, 9, 10,
                                      11, 12, 13, 14, 15);
    v8f zeroc = {};
    v8f o[4];
#pragma unroll
    for (int nt = 0; nt < 4; ++nt)
      o[nt] = wmma16(ah, fetch_bfrag(sW2f[t], nt), zeroc);

    // weh = (OUT+b2)*hx[sender], scatter-add at receiver (type known)
#pragma unroll
    for (int v = 0; v < 8; ++v) {
      const int idx = base + v + 8 * hi;
      if (idx < ct) {
        const int e = lst[idx];
        const int s = senders[e];
        const int rcv = receivers[e];
#pragma unroll
        for (int nt = 0; nt < 4; ++nt) {
          const int n = nt * 16 + n0;
          const float w = o[nt][v] + sB2[t][n];
          atomicAdd(&z[(size_t)t * ZSTRIDE + (size_t)rcv * K + n],
                    w * hx[(size_t)s * K + n]);
        }
      }
    }
    wave_lds_sync();  // Hw reads done before next iteration overwrites
  }
}

// out = elec + sum_t ( z[t][elec nodes] @ gW_t + gb_t )
__global__ void __launch_bounds__(128) out_kernel(
    const float* __restrict__ elec, const float* __restrict__ z,
    const float* __restrict__ gW_same, const float* __restrict__ gb_same,
    const float* __restrict__ gW_anti, const float* __restrict__ gb_anti,
    const float* __restrict__ gW_n,    const float* __restrict__ gb_n,
    float* __restrict__ out) {
  __shared__ alignas(32) _Float16 sG[3][2 * 8 * 512];  // 2 K-steps x 8 N-tiles
  const float* gW[3] = {gW_same, gW_anti, gW_n};
#pragma unroll
  for (int t = 0; t < 3; ++t) stage_bfrags(gW[t], D, 2, 8, sG[t]);
  __syncthreads();

  const int lane = threadIdx.x & 31;
  const int wid  = threadIdx.x >> 5;
  const int wt = blockIdx.x * 4 + wid;   // 16 elec rows
  const int r0 = wt * 16;
  // rows r0..r0+15 share one batch -> contiguous node range in z
  const int node0 = (r0 >> 6) * (NN + NE) + NN + (r0 & 63);
  v8f zero = {};
  v8f acc[8];
#pragma unroll
  for (int nt = 0; nt < 8; ++nt) acc[nt] = zero;

#pragma unroll 1
  for (int t = 0; t < 3; ++t) {
    const float* zt = z + (size_t)t * ZSTRIDE + (size_t)node0 * K;
#pragma unroll
    for (int kt = 0; kt < 2; ++kt) {     // K=64 -> 2 steps of 32
      v16h az = load_a_vec(zt + kt * 32, K);
#pragma unroll
      for (int nt = 0; nt < 8; ++nt)     // D=128 -> 8 N-tiles
        acc[nt] = wmma16(az, fetch_bfrag(sG[t], kt * 8 + nt), acc[nt]);
    }
  }
  const int n0 = lane & 15, hi = lane >> 4;
#pragma unroll
  for (int nt = 0; nt < 8; ++nt) {
    const int n = nt * 16 + n0;
    const float bias = gb_same[n] + gb_anti[n] + gb_n[n];
#pragma unroll
    for (int v = 0; v < 8; ++v) {
      const int r = r0 + v + 8 * hi;
      out[(size_t)r * D + n] = elec[(size_t)r * D + n] + acc[nt][v] + bias;
    }
  }
}

extern "C" void kernel_launch(void* const* d_in, const int* in_sizes, int n_in,
                              void* d_out, int out_size, void* d_ws,
                              size_t ws_size, hipStream_t stream) {
  const float* nuc  = (const float*)d_in[0];
  const float* elec = (const float*)d_in[1];
  const float* dist = (const float*)d_in[2];
  const float* w_same_W1 = (const float*)d_in[3];
  const float* w_same_b1 = (const float*)d_in[4];
  const float* w_same_W2 = (const float*)d_in[5];
  const float* w_same_b2 = (const float*)d_in[6];
  const float* g_same_W  = (const float*)d_in[7];
  const float* g_same_b  = (const float*)d_in[8];
  const float* w_anti_W1 = (const float*)d_in[9];
  const float* w_anti_b1 = (const float*)d_in[10];
  const float* w_anti_W2 = (const float*)d_in[11];
  const float* w_anti_b2 = (const float*)d_in[12];
  const float* g_anti_W  = (const float*)d_in[13];
  const float* g_anti_b  = (const float*)d_in[14];
  const float* w_n_W1 = (const float*)d_in[15];
  const float* w_n_b1 = (const float*)d_in[16];
  const float* w_n_W2 = (const float*)d_in[17];
  const float* w_n_b2 = (const float*)d_in[18];
  const float* g_n_W  = (const float*)d_in[19];
  const float* g_n_b  = (const float*)d_in[20];
  const float* h_W = (const float*)d_in[21];
  const float* h_b = (const float*)d_in[22];
  const int* e_typ     = (const int*)d_in[23];
  const int* senders   = (const int*)d_in[24];
  const int* receivers = (const int*)d_in[25];
  const int E = in_sizes[23];

  float* hx   = (float*)d_ws;                    // 5120*64 f32
  float* zbuf = hx + (size_t)NNODES * K;         // 3*5120*64 f32
  int*   cnt  = (int*)(zbuf + 3 * (size_t)ZSTRIDE);  // 4 ints (zeroed below)
  int*   lists = cnt + 4;                        // 3*E ints

  zero_kernel<<<512, 256, 0, stream>>>(zbuf, 3 * ZSTRIDE + 4);
  compact_kernel<<<1024, 256, 0, stream>>>(e_typ, E, cnt, lists);
  hx_kernel<<<64 + 16, 128, 0, stream>>>(nuc, elec, h_W, h_b, hx);
  edge_kernel<<<512, 128, 0, stream>>>(
      dist, w_same_W1, w_same_b1, w_same_W2, w_same_b2, w_anti_W1, w_anti_b1,
      w_anti_W2, w_anti_b2, w_n_W1, w_n_b1, w_n_W2, w_n_b2, senders, receivers,
      hx, zbuf, cnt, lists, E);
  out_kernel<<<NELEC / 64, 128, 0, stream>>>(elec, zbuf, g_same_W, g_same_b,
                                             g_anti_W, g_anti_b, g_n_W, g_n_b,
                                             (float*)d_out);
}
